// SelfAttention_3143916061349
// MI455X (gfx1250) — matrix operand order repaired and checked by reading, not compiled, not run on hardware
//
#include <hip/hip_runtime.h>

// ---------------------------------------------------------------------------
// Self-attention for MI455X (gfx1250): bf16 WMMA (v_wmma_f32_16x16x32_bf16),
// flash-attention style with async global->LDS staging (ASYNCcnt path),
// fp32 accumulate, fp32 in/out.  B=4, S=4096, D=256.
// ---------------------------------------------------------------------------

#define B_  4
#define S_  4096
#define D_  256

typedef __attribute__((ext_vector_type(16))) __bf16 v16bf;
typedef __attribute__((ext_vector_type(8)))  float  v8f;
typedef int v4i __attribute__((ext_vector_type(4)));
typedef __attribute__((address_space(1))) v4i as1_v4i;   // global int4
typedef __attribute__((address_space(3))) v4i as3_v4i;   // LDS int4

#if __has_builtin(__builtin_amdgcn_global_load_async_to_lds_b128)
#define HAVE_ASYNC_LDS 1
#else
#define HAVE_ASYNC_LDS 0
#endif

// Copy 16 bytes global -> LDS. Async (ASYNCcnt) when the toolchain exposes
// the gfx1250 async-LDS builtin; plain load+store otherwise.
__device__ __forceinline__ void cp_g2l_16B(unsigned short* ldst, const unsigned short* gsrc) {
#if HAVE_ASYNC_LDS
    __builtin_amdgcn_global_load_async_to_lds_b128(
        (as1_v4i*)gsrc, (as3_v4i*)ldst, /*offset=*/0, /*cpol=*/0);
#else
    *(uint4*)ldst = *(const uint4*)gsrc;
#endif
}

// Wait for this wave's outstanding async-LDS loads, then block barrier.
__device__ __forceinline__ void async_join() {
#if HAVE_ASYNC_LDS
#if __has_builtin(__builtin_amdgcn_s_wait_asynccnt)
    __builtin_amdgcn_s_wait_asynccnt(0);
#else
    asm volatile("s_wait_asynccnt 0" ::: "memory");
#endif
#endif
    __syncthreads();
}

// Intra-wave LDS ordering (store -> load of the per-wave P tile).
__device__ __forceinline__ void lds_fence_wave() {
#if __has_builtin(__builtin_amdgcn_s_wait_dscnt)
    __builtin_amdgcn_s_wait_dscnt(0);
#else
    asm volatile("s_wait_dscnt 0" ::: "memory");
#endif
}

__device__ __forceinline__ unsigned short f32_to_bf16(float f) {
    unsigned int u = __builtin_bit_cast(unsigned int, f);
    u += 0x7fffu + ((u >> 16) & 1u);            // round-to-nearest-even
    return (unsigned short)(u >> 16);
}

__device__ __forceinline__ unsigned int pack2_bf16(float a, float b) {
    unsigned int ua = __builtin_bit_cast(unsigned int, a);
    unsigned int ub = __builtin_bit_cast(unsigned int, b);
    ua += 0x7fffu + ((ua >> 16) & 1u);
    ub += 0x7fffu + ((ub >> 16) & 1u);
    return (ua >> 16) | (ub & 0xffff0000u);
}

// WMMA wrapper: D = A(16x32 bf16) * B(32x16 bf16) + C(f32)
__device__ __forceinline__ v8f wmma_bf16(v16bf a, v16bf b, v8f c) {
    return __builtin_amdgcn_wmma_f32_16x16x32_bf16(
        false, a, false, b, (short)0, c, false, false);
}

// A-fragment from f32 row (convert to bf16 on the fly).
// 16-bit A 16x32: lane holds row m; contiguous k-chunk
// [kbase+kh*16, +16), kh = lane>>4 (matches B fragment order).
__device__ __forceinline__ v16bf load_afrag_f32(const float* rowbase, int kbase, int kh) {
    const float4* p0 = (const float4*)(rowbase + kbase + kh * 16);
    const float4* p1 = (const float4*)(rowbase + kbase + kh * 16 + 8);
    float4 f0 = p0[0], f1 = p0[1], f2 = p1[0], f3 = p1[1];
    union { v16bf v; unsigned int u[8]; } r;
    r.u[0] = pack2_bf16(f0.x, f0.y); r.u[1] = pack2_bf16(f0.z, f0.w);
    r.u[2] = pack2_bf16(f1.x, f1.y); r.u[3] = pack2_bf16(f1.z, f1.w);
    r.u[4] = pack2_bf16(f2.x, f2.y); r.u[5] = pack2_bf16(f2.z, f2.w);
    r.u[6] = pack2_bf16(f3.x, f3.y); r.u[7] = pack2_bf16(f3.z, f3.w);
    return r.v;
}

// A-fragment from bf16 row-major storage (two 16B loads; works for LDS too).
__device__ __forceinline__ v16bf load_afrag_bf16(const unsigned short* rowbase, int kbase, int kh) {
    union { v16bf v; uint4 u[2]; } r;
    r.u[0] = *(const uint4*)(rowbase + kbase + kh * 16);
    r.u[1] = *(const uint4*)(rowbase + kbase + kh * 16 + 8);
    return r.v;
}

// B-fragment: lane holds column n, 16 contiguous bf16 (32B), k fast axis.
__device__ __forceinline__ v16bf load_bfrag_bf16(const unsigned short* p) {
    union { v16bf v; uint4 u[2]; } r;
    const uint4* q = (const uint4*)p;
    r.u[0] = q[0]; r.u[1] = q[1];
    return r.v;
}

// ---------------------------------------------------------------------------
// Kernel 0: transpose + convert weights: Wt[mat][n][k] = bf16(W[mat][k][n])
// ---------------------------------------------------------------------------
__global__ __launch_bounds__(256) void wt_convert_kernel(
    const float* __restrict__ Wq, const float* __restrict__ Wk,
    const float* __restrict__ Wv, unsigned short* __restrict__ Wt)
{
    int idx = blockIdx.x * 256 + threadIdx.x;      // 3 * 256 * 256 total
    int mat = idx >> 16;
    int r   = idx & 65535;
    int k   = r >> 8;
    int n   = r & 255;
    const float* W = (mat == 0) ? Wq : ((mat == 1) ? Wk : Wv);
    Wt[(size_t)mat * 65536 + n * 256 + k] = f32_to_bf16(W[k * 256 + n]);
}

// ---------------------------------------------------------------------------
// Kernel 1: QKV projection. One wave = one 16-row block of one matrix.
// Q, K stored row-major bf16 [b][s][d]; V stored transposed bf16 [b][d][s].
// ---------------------------------------------------------------------------
__global__ __launch_bounds__(256) void qkv_proj_kernel(
    const float* __restrict__ x, const unsigned short* __restrict__ Wt,
    const float* __restrict__ bq, const float* __restrict__ bk,
    const float* __restrict__ bv,
    unsigned short* __restrict__ Qb, unsigned short* __restrict__ Kb,
    unsigned short* __restrict__ Vt)
{
    int lane = threadIdx.x & 31;
    int wid  = blockIdx.x * (blockDim.x >> 5) + (threadIdx.x >> 5);
    int mat  = wid >> 10;                 // 0=Q 1=K 2=V
    int rb   = wid & 1023;                // 16-row block index over B*S
    int b    = rb >> 8;
    int s0   = (rb & 255) << 4;
    int m    = lane & 15;
    int kh   = lane >> 4;

    const unsigned short* W    = Wt + (size_t)mat * 65536;
    const float*          bias = (mat == 0) ? bq : ((mat == 1) ? bk : bv);

    const float* xrow = x + ((size_t)(b * S_) + s0 + m) * D_;
    v16bf a[8];
#pragma unroll
    for (int kc = 0; kc < 8; ++kc) a[kc] = load_afrag_f32(xrow, kc * 32, kh);

    for (int nt = 0; nt < 16; ++nt) {
        int ncol = nt * 16 + m;
        const unsigned short* wrow = W + (size_t)ncol * 256;
        v8f c = {};
#pragma unroll
        for (int kc = 0; kc < 8; ++kc) {
            v16bf bf = load_bfrag_bf16(wrow + kc * 32 + kh * 16);
            c = wmma_bf16(a[kc], bf, c);
        }
        float bb = bias[ncol];
#pragma unroll
        for (int r = 0; r < 8; ++r) c[r] += bb;

        if (mat == 2) {
            // transposed store: Vt[b][ncol][s0 + kh*8 + r], contiguous in t
            unsigned short* vp = Vt + ((size_t)(b * D_) + ncol) * S_ + s0 + kh * 8;
            union { uint4 q; unsigned int u[4]; } pk;
            pk.u[0] = pack2_bf16(c[0], c[1]); pk.u[1] = pack2_bf16(c[2], c[3]);
            pk.u[2] = pack2_bf16(c[4], c[5]); pk.u[3] = pack2_bf16(c[6], c[7]);
            *(uint4*)vp = pk.q;
        } else {
            unsigned short* dst = (mat == 0) ? Qb : Kb;
#pragma unroll
            for (int r = 0; r < 8; ++r) {
                size_t idx = ((size_t)(b * S_) + s0 + r + 8 * kh) * D_ + ncol;
                dst[idx] = f32_to_bf16(c[r]);
            }
        }
    }
}

// ---------------------------------------------------------------------------
// Kernel 2: flash attention (unscaled, matches reference) + residual.
// 4 waves / block = 64 queries; K/V tiles (32 keys) double-buffered in LDS,
// staged with async global->LDS loads; online softmax; P transposed through
// per-wave LDS tile; 16 d-out WMMA accumulators per wave.
// ---------------------------------------------------------------------------
__global__ __launch_bounds__(128) void attn_kernel(
    const float* __restrict__ x,
    const unsigned short* __restrict__ Qb, const unsigned short* __restrict__ Kb,
    const unsigned short* __restrict__ Vt, float* __restrict__ out)
{
    __shared__ unsigned short Ql[64 * 256];        // 32 KB: Q rows of this block
    __shared__ unsigned short Kl[2][32 * 256];     // 2 x 16 KB: K tile [t][d]
    __shared__ unsigned short Vl[2][256 * 32];     // 2 x 16 KB: V tile [d][t]
    __shared__ unsigned short Pw[4][16 * 32];      // 4 x 1 KB: per-wave P tile

    const int tid  = threadIdx.x;
    const int warp = tid >> 5;
    const int lane = tid & 31;
    const int m    = lane & 15;
    const int kh   = lane >> 4;
    const int b    = blockIdx.x >> 6;              // 64 tokens/block, 4096/batch
    const int tok0 = blockIdx.x * 64;              // first (global) token of block
    const int NIT  = S_ / 32;

    // ---- stage Q (64 rows x 256, contiguous 32 KB) + first K/V tile ----
    {
        const unsigned short* qsrc = Qb + (size_t)tok0 * D_;
#pragma unroll
        for (int u = 0; u < 16; ++u) {
            int unit = tid + u * 128;
            cp_g2l_16B(&Ql[unit * 8], qsrc + unit * 8);
        }
    }
    {
        // K tile: 32 rows x 512B, contiguous block of Kb
        const unsigned short* ksrc = Kb + (size_t)(b * S_) * D_;
#pragma unroll
        for (int u = 0; u < 16; ++u) {
            int unit = tid + u * 128;
            cp_g2l_16B(&Kl[0][unit * 8], ksrc + unit * 8);
        }
        // V tile: 256 rows of 64B from Vt[b][d][0..32)
#pragma unroll
        for (int u = 0; u < 8; ++u) {
            int unit = tid + u * 128;              // 0..1023
            int d    = unit >> 2;
            int part = unit & 3;
            cp_g2l_16B(&Vl[0][unit * 8],
                       Vt + ((size_t)(b * D_) + d) * S_ + part * 8);
        }
    }
    async_join();

    v8f O[16];
    v8f zero = {};
#pragma unroll
    for (int dt = 0; dt < 16; ++dt) O[dt] = zero;
    float rowM[8], rowL[8];
#pragma unroll
    for (int r = 0; r < 8; ++r) { rowM[r] = -3.0e38f; rowL[r] = 0.0f; }

    const unsigned short* qrow = &Ql[(warp * 16 + m) * 256];

    for (int it = 0; it < NIT; ++it) {
        const int cur = it & 1;

        // ---- stage next K/V tile into the other buffer (async) ----
        if (it + 1 < NIT) {
            int t0n = (it + 1) * 32;
            const unsigned short* ksrc = Kb + ((size_t)(b * S_) + t0n) * D_;
#pragma unroll
            for (int u = 0; u < 16; ++u) {
                int unit = tid + u * 128;
                cp_g2l_16B(&Kl[cur ^ 1][unit * 8], ksrc + unit * 8);
            }
#pragma unroll
            for (int u = 0; u < 8; ++u) {
                int unit = tid + u * 128;
                int d    = unit >> 2;
                int part = unit & 3;
                cp_g2l_16B(&Vl[cur ^ 1][unit * 8],
                           Vt + ((size_t)(b * D_) + d) * S_ + t0n + part * 8);
            }
        }

        const unsigned short* Kc = Kl[cur];
        const unsigned short* Vc = Vl[cur];

        // ---- scores: two 16x16 tiles over 32 keys, K-dim = 256 ----
        v8f c0 = zero, c1 = zero;
#pragma unroll
        for (int kc = 0; kc < 8; ++kc) {
            v16bf aq  = load_afrag_bf16(qrow, kc * 32, kh);
            v16bf bk0 = load_bfrag_bf16(Kc + (size_t)m * 256 + kc * 32 + kh * 16);
            v16bf bk1 = load_bfrag_bf16(Kc + (size_t)(16 + m) * 256 + kc * 32 + kh * 16);
            c0 = wmma_bf16(aq, bk0, c0);
            c1 = wmma_bf16(aq, bk1, c1);
        }

        // ---- online softmax (16-lane half-group reductions) ----
        float corr[8];
#pragma unroll
        for (int r = 0; r < 8; ++r) {
            float mx = fmaxf(c0[r], c1[r]);
#pragma unroll
            for (int off = 1; off < 16; off <<= 1)
                mx = fmaxf(mx, __shfl_xor(mx, off, 32));
            float nm = fmaxf(rowM[r], mx);
            corr[r]  = __expf(rowM[r] - nm);
            rowM[r]  = nm;
            float p0 = __expf(c0[r] - nm);
            float p1 = __expf(c1[r] - nm);
            c0[r] = p0; c1[r] = p1;
            float s = p0 + p1;
#pragma unroll
            for (int off = 1; off < 16; off <<= 1)
                s += __shfl_xor(s, off, 32);
            rowL[r] = rowL[r] * corr[r] + s;
        }

        // ---- transpose P via this wave's private LDS tile ----
        unsigned short* P = Pw[warp];
#pragma unroll
        for (int r = 0; r < 8; ++r) {
            int row = r + 8 * kh;
            P[row * 32 + m]      = f32_to_bf16(c0[r]);
            P[row * 32 + 16 + m] = f32_to_bf16(c1[r]);
        }
        lds_fence_wave();
        v16bf pf;
        {
            union { v16bf v; uint4 u[2]; } rr;
            rr.u[0] = *(const uint4*)&P[m * 32 + kh * 16];
            rr.u[1] = *(const uint4*)&P[m * 32 + kh * 16 + 8];
            pf = rr.v;
        }

        // ---- rescale accumulators, O += P @ V over 16 d-out tiles ----
#pragma unroll
        for (int dt = 0; dt < 16; ++dt) {
#pragma unroll
            for (int r = 0; r < 8; ++r) O[dt][r] *= corr[r];
            v16bf bv = load_bfrag_bf16(Vc + (size_t)(dt * 16 + m) * 32 + kh * 16);
            O[dt] = wmma_bf16(pf, bv, O[dt]);
        }

        // next buffer ready + everyone done with current buffer
        async_join();
    }

    // ---- epilogue: normalize, residual, store fp32 ----
#pragma unroll
    for (int dt = 0; dt < 16; ++dt) {
#pragma unroll
        for (int r = 0; r < 8; ++r) {
            size_t row = (size_t)tok0 + warp * 16 + r + 8 * kh;
            size_t idx = row * D_ + dt * 16 + m;
            out[idx] = O[dt][r] / rowL[r] + x[idx];
        }
    }
}

// ---------------------------------------------------------------------------
// Host launcher
// ---------------------------------------------------------------------------
extern "C" void kernel_launch(void* const* d_in, const int* in_sizes, int n_in,
                              void* d_out, int out_size, void* d_ws, size_t ws_size,
                              hipStream_t stream) {
    const float* x  = (const float*)d_in[0];
    const float* Wq = (const float*)d_in[1];
    const float* bq = (const float*)d_in[2];
    const float* Wk = (const float*)d_in[3];
    const float* bk = (const float*)d_in[4];
    const float* Wv = (const float*)d_in[5];
    const float* bv = (const float*)d_in[6];
    float* out = (float*)d_out;

    const size_t NTOK = (size_t)B_ * S_;          // 16384
    unsigned short* Qb = (unsigned short*)d_ws;   // [B][S][D] bf16
    unsigned short* Kb = Qb + NTOK * D_;          // [B][S][D] bf16
    unsigned short* Vt = Kb + NTOK * D_;          // [B][D][S] bf16 (transposed)
    unsigned short* Wt = Vt + NTOK * D_;          // 3 x [n][k] bf16

    wt_convert_kernel<<<768, 256, 0, stream>>>(Wq, Wk, Wv, Wt);
    qkv_proj_kernel<<<384, 256, 0, stream>>>(x, Wt, bq, bk, bv, Qb, Kb, Vt);
    // 4 waves (64 queries) per block: B*S/64 = 256 blocks
    attn_kernel<<<256, 128, 0, stream>>>(x, Qb, Kb, Vt, out);
}
